// SpatialRelationLayer_41308995453143
// MI455X (gfx1250) — matrix operand order repaired and checked by reading, not compiled, not run on hardware
//
#include <hip/hip_runtime.h>
#include <hip/hip_bf16.h>

// ---------------------------------------------------------------------------
// SpatialRelationLayer for MI455X (gfx1250, wave32, WMMA).
// Fused per-wave MLP tiles using v_wmma_f32_16x16x32_f16 (f16 in, f32 acc),
// hidden width padded 132 -> 160. Weights pre-swizzled to B-fragment layout
// (one coalesced 32B/lane load per fragment). M-blocked 2 A-tiles per wave
// so every B-fragment load feeds two WMMAs (halves L2 weight traffic and
// hides the load->wmma wait).
// ---------------------------------------------------------------------------

typedef __attribute__((ext_vector_type(16))) _Float16 v16h;
typedef __attribute__((ext_vector_type(8)))  _Float16 v8h;
typedef __attribute__((ext_vector_type(8)))  float    v8f;

union FragA { v16h v; v8h h[2]; };

#define HP   160   // padded hidden width (132 -> 160, multiple of 32)
#define NT_H 10    // hidden N-tiles (160/16)
#define KT_H 5     // hidden K-tiles (160/32)
#define NT_O 4     // output N-tiles (64/16)

// ---------------------------------------------------------------------------
// Max reductions: Qmax[b,d] = max_n Q[b,n,d];
// Rmax0[b,n,d] = max_i R[b,i,n,d]; Rmax1[b,n,d] = max_j R[b,n,j,d]
// ---------------------------------------------------------------------------
__global__ void qmax_kernel(const float* __restrict__ Q, float* __restrict__ qmax) {
    int t = blockIdx.x * blockDim.x + threadIdx.x;
    if (t >= 8 * 128) return;
    int b = t >> 7, d = t & 127;
    float m = -3.402823466e38f;
    for (int n = 0; n < 128; ++n) m = fmaxf(m, Q[(b * 128 + n) * 128 + d]);
    qmax[t] = m;
}

__global__ void rmax_kernel(const float* __restrict__ R,
                            float* __restrict__ rmax0, float* __restrict__ rmax1) {
    int t = blockIdx.x * blockDim.x + threadIdx.x;
    if (t >= 8 * 128 * 64) return;
    int b = t >> 13, n = (t >> 6) & 127, d = t & 63;
    float m0 = -3.402823466e38f, m1 = -3.402823466e38f;
    for (int i = 0; i < 128; ++i) {
        m0 = fmaxf(m0, R[(((b * 128 + i) * 128) + n) * 64 + d]);  // max over axis 1
        m1 = fmaxf(m1, R[(((b * 128 + n) * 128) + i) * 64 + d]);  // max over axis 2
    }
    rmax0[t] = m0;
    rmax1[t] = m1;
}

// ---------------------------------------------------------------------------
// Weight pre-swizzle: f32 row-major [Ksrc, Nsrc] -> f16 B-fragment blocks.
// Block (nt, kt) = 32 lanes x 16 halves; lane l, half h holds
//   W[kt*32 + (l>>4)*8 + (h&7) + (h>>3)*16][nt*16 + (l&15)]   (zero outside source extent).
// ---------------------------------------------------------------------------
__global__ void swizzle_kernel(const float* __restrict__ src, _Float16* __restrict__ dst,
                               int Ksrc, int Nsrc, int Kpad, int Npad) {
    int t = blockIdx.x * blockDim.x + threadIdx.x;
    if (t >= Kpad * Npad) return;
    int blk  = t >> 9;
    int lane = (t >> 4) & 31;
    int h    = t & 15;
    int nkt  = Kpad >> 5;
    int kt   = blk % nkt;
    int nt   = blk / nkt;
    int K = kt * 32 + ((lane >> 4) << 3) + (h & 7) + ((h >> 3) << 4);
    int N = nt * 16 + (lane & 15);
    float v = (K < Ksrc && N < Nsrc) ? src[K * Nsrc + N] : 0.0f;
    dst[t] = (_Float16)v;
}

// ---------------------------------------------------------------------------
// Fused MLP: per-wave MT 16-row A-tiles, WMMA f16 -> f32.
// BRANCH 0: P  (M=8,     KIN=256: [Qmax | P])
// BRANCH 1: Q  (M=1024,  KIN=384: [P | Q | Rmax0 | Rmax1])
// BRANCH 2: R  (M=131072,KIN=320: [R(b,i,j) | Q(b,j) | Q(b,i)])
// ---------------------------------------------------------------------------
template <int BRANCH, int KIN, int MT>
__global__ __launch_bounds__(128) void mlp_fused(
    const float* __restrict__ P, const float* __restrict__ Q, const float* __restrict__ R,
    const float* __restrict__ qmax, const float* __restrict__ rmax0, const float* __restrict__ rmax1,
    const _Float16* __restrict__ w1s, const float* __restrict__ b1,
    const _Float16* __restrict__ w2s, const float* __restrict__ b2,
    const _Float16* __restrict__ w3s, const float* __restrict__ b3,
    float* __restrict__ out, int M)
{
    constexpr int KT1  = KIN / 32;
    constexpr int ROWS = 16 * MT;
    constexpr int XSZ  = (ROWS * KIN > 2 * ROWS * HP) ? (ROWS * KIN) : (2 * ROWS * HP);
    __shared__ _Float16 smem[4][XSZ];

    const int wave = threadIdx.x >> 5;
    const int lane = threadIdx.x & 31;
    const int hi   = lane >> 4;   // half-wave group
    const int lo   = lane & 15;
    const int m0   = (blockIdx.x * 4 + wave) * ROWS;

    _Float16* Xs = smem[wave];               // input tile [ROWS][KIN] (f16)
    _Float16* H1 = smem[wave];               // layer-1 act [ROWS][HP] (reuses X region)
    _Float16* H2 = smem[wave] + ROWS * HP;   // layer-2 act [ROWS][HP]

    // ---- stage input rows (f32 -> f16), zero-fill past M ----
    for (int idx = lane; idx < ROWS * KIN; idx += 32) {
        const int row = idx / KIN;
        const int k   = idx - row * KIN;
        const int gm  = m0 + row;
        float v = 0.0f;
        if (gm < M) {
            if constexpr (BRANCH == 0) {
                const int b = gm;
                v = (k < 128) ? qmax[b * 128 + k] : P[b * 128 + (k - 128)];
            } else if constexpr (BRANCH == 1) {
                const int b = gm >> 7, n = gm & 127;
                if      (k < 128) v = P[b * 128 + k];
                else if (k < 256) v = Q[(b * 128 + n) * 128 + (k - 128)];
                else if (k < 320) v = rmax0[(b * 128 + n) * 64 + (k - 256)];
                else              v = rmax1[(b * 128 + n) * 64 + (k - 320)];
            } else {
                const int b = gm >> 14, i = (gm >> 7) & 127, j = gm & 127;
                if      (k < 64)  v = R[(((b * 128 + i) * 128) + j) * 64 + k];
                else if (k < 192) v = Q[(b * 128 + j) * 128 + (k - 64)];
                else              v = Q[(b * 128 + i) * 128 + (k - 192)];
            }
        }
        Xs[row * KIN + k] = (_Float16)v;
    }
    __syncthreads();

    // ---- layer 1: [ROWS,KIN] x [KIN,HP] ----
    v8f acc1[MT][NT_H] = {};
    #pragma unroll
    for (int kt = 0; kt < KT1; ++kt) {
        FragA a[MT];
        #pragma unroll
        for (int t = 0; t < MT; ++t) {
            const _Float16* arow = Xs + (t * 16 + lo) * KIN + kt * 32 + hi * 8;
            a[t].h[0] = *(const v8h*)(arow);
            a[t].h[1] = *(const v8h*)(arow + 16);
        }
        #pragma unroll
        for (int nt = 0; nt < NT_H; ++nt) {
            v16h bf = *(const v16h*)(w1s + ((nt * KT1 + kt) << 9) + (lane << 4));
            #pragma unroll
            for (int t = 0; t < MT; ++t)
                acc1[t][nt] = __builtin_amdgcn_wmma_f32_16x16x32_f16(
                    false, a[t].v, false, bf, (short)0, acc1[t][nt], false, false);
        }
    }
    // bias + ReLU -> H1 (f16)
    #pragma unroll
    for (int t = 0; t < MT; ++t) {
        #pragma unroll
        for (int nt = 0; nt < NT_H; ++nt) {
            #pragma unroll
            for (int r = 0; r < 8; ++r) {
                const int row = t * 16 + hi * 8 + r;
                const int col = nt * 16 + lo;
                float x = acc1[t][nt][r] + ((col < 132) ? b1[col] : 0.0f);
                H1[row * HP + col] = (_Float16)fmaxf(x, 0.0f);
            }
        }
    }
    __syncthreads();

    // ---- layer 2: [ROWS,HP] x [HP,HP] ----
    v8f acc2[MT][NT_H] = {};
    #pragma unroll
    for (int kt = 0; kt < KT_H; ++kt) {
        FragA a[MT];
        #pragma unroll
        for (int t = 0; t < MT; ++t) {
            const _Float16* arow = H1 + (t * 16 + lo) * HP + kt * 32 + hi * 8;
            a[t].h[0] = *(const v8h*)(arow);
            a[t].h[1] = *(const v8h*)(arow + 16);
        }
        #pragma unroll
        for (int nt = 0; nt < NT_H; ++nt) {
            v16h bf = *(const v16h*)(w2s + ((nt * KT_H + kt) << 9) + (lane << 4));
            #pragma unroll
            for (int t = 0; t < MT; ++t)
                acc2[t][nt] = __builtin_amdgcn_wmma_f32_16x16x32_f16(
                    false, a[t].v, false, bf, (short)0, acc2[t][nt], false, false);
        }
    }
    #pragma unroll
    for (int t = 0; t < MT; ++t) {
        #pragma unroll
        for (int nt = 0; nt < NT_H; ++nt) {
            #pragma unroll
            for (int r = 0; r < 8; ++r) {
                const int row = t * 16 + hi * 8 + r;
                const int col = nt * 16 + lo;
                float x = acc2[t][nt][r] + ((col < 132) ? b2[col] : 0.0f);
                H2[row * HP + col] = (_Float16)fmaxf(x, 0.0f);
            }
        }
    }
    __syncthreads();

    // ---- layer 3: [ROWS,HP] x [HP,64], sigmoid(5x) epilogue ----
    v8f acc3[MT][NT_O] = {};
    #pragma unroll
    for (int kt = 0; kt < KT_H; ++kt) {
        FragA a[MT];
        #pragma unroll
        for (int t = 0; t < MT; ++t) {
            const _Float16* arow = H2 + (t * 16 + lo) * HP + kt * 32 + hi * 8;
            a[t].h[0] = *(const v8h*)(arow);
            a[t].h[1] = *(const v8h*)(arow + 16);
        }
        #pragma unroll
        for (int nt = 0; nt < NT_O; ++nt) {
            v16h bf = *(const v16h*)(w3s + ((nt * KT_H + kt) << 9) + (lane << 4));
            #pragma unroll
            for (int t = 0; t < MT; ++t)
                acc3[t][nt] = __builtin_amdgcn_wmma_f32_16x16x32_f16(
                    false, a[t].v, false, bf, (short)0, acc3[t][nt], false, false);
        }
    }
    #pragma unroll
    for (int t = 0; t < MT; ++t) {
        #pragma unroll
        for (int nt = 0; nt < NT_O; ++nt) {
            #pragma unroll
            for (int r = 0; r < 8; ++r) {
                const int gm  = m0 + t * 16 + hi * 8 + r;
                const int col = nt * 16 + lo;
                if (gm < M) {
                    float x = (acc3[t][nt][r] + b3[col]) * 5.0f;
                    out[gm * 64 + col] = 1.0f / (1.0f + __expf(-x));
                }
            }
        }
    }
}

// ---------------------------------------------------------------------------
extern "C" void kernel_launch(void* const* d_in, const int* in_sizes, int n_in,
                              void* d_out, int out_size, void* d_ws, size_t ws_size,
                              hipStream_t stream) {
    const float* P = (const float*)d_in[0];
    const float* Q = (const float*)d_in[1];
    const float* R = (const float*)d_in[2];

    // weights per branch: [w1, b1, w2, b2, w3, b3]; branches p=0, q=1, r=2
    const float* W[3][6];
    for (int br = 0; br < 3; ++br)
        for (int j = 0; j < 6; ++j)
            W[br][j] = (const float*)d_in[3 + br * 6 + j];

    const int KIN[3] = {256, 384, 320};

    // -- workspace carving (256B aligned); total ~1.05 MB --
    char*  ws  = (char*)d_ws;
    size_t off = 0;
    auto carve = [&](size_t bytes) -> void* {
        void* p = ws + off;
        off = (off + bytes + 255) & ~(size_t)255;
        return p;
    };
    float* qmax  = (float*)carve(8 * 128 * sizeof(float));
    float* rmax0 = (float*)carve(8 * 128 * 64 * sizeof(float));
    float* rmax1 = (float*)carve(8 * 128 * 64 * sizeof(float));
    _Float16* w1s[3]; _Float16* w2s[3]; _Float16* w3s[3];
    for (int br = 0; br < 3; ++br) {
        w1s[br] = (_Float16*)carve((size_t)KIN[br] * HP * 2);
        w2s[br] = (_Float16*)carve((size_t)HP * HP * 2);
        w3s[br] = (_Float16*)carve((size_t)HP * 64 * 2);
    }

    // -- reductions --
    qmax_kernel<<<(8 * 128 + 255) / 256, 256, 0, stream>>>(Q, qmax);
    rmax_kernel<<<(8 * 128 * 64 + 255) / 256, 256, 0, stream>>>(R, rmax0, rmax1);

    // -- weight pre-swizzle (zero-padded to WMMA-friendly extents) --
    for (int br = 0; br < 3; ++br) {
        int t1 = KIN[br] * HP;
        swizzle_kernel<<<(t1 + 255) / 256, 256, 0, stream>>>(W[br][0], w1s[br], KIN[br], 132, KIN[br], HP);
        int t2 = HP * HP;
        swizzle_kernel<<<(t2 + 255) / 256, 256, 0, stream>>>(W[br][2], w2s[br], 132, 132, HP, HP);
        int t3 = HP * 64;
        swizzle_kernel<<<(t3 + 255) / 256, 256, 0, stream>>>(W[br][4], w3s[br], 132, 64, HP, 64);
    }

    // -- fused MLPs (outputs concatenated: P_out[512] | Q_out[65536] | R_out[8388608]) --
    float* outP = (float*)d_out;
    float* outQ = outP + 512;
    float* outR = outQ + 8 * 128 * 64;

    // P branch: M=8 -> one block, single 16-row tile per wave
    mlp_fused<0, 256, 1><<<1, 128, 0, stream>>>(
        P, Q, R, qmax, rmax0, rmax1,
        w1s[0], W[0][1], w2s[0], W[0][3], w3s[0], W[0][5], outP, 8);

    // Q branch: M=1024, 32 rows/wave -> 8 blocks
    mlp_fused<1, 384, 2><<<8, 128, 0, stream>>>(
        P, Q, R, qmax, rmax0, rmax1,
        w1s[1], W[1][1], w2s[1], W[1][3], w3s[1], W[1][5], outQ, 8 * 128);

    // R branch (dominant): M=131072, 32 rows/wave -> 1024 blocks
    mlp_fused<2, 320, 2><<<1024, 128, 0, stream>>>(
        P, Q, R, qmax, rmax0, rmax1,
        w1s[2], W[2][1], w2s[2], W[2][3], w3s[2], W[2][5], outR, 8 * 128 * 128);
}
